// SEGNO_80315888435714
// MI455X (gfx1250) — compile-verified, matrix operand order checked by reading, not run on hardware
//
#include <hip/hip_runtime.h>

// ---------------- problem constants ----------------
#define NN 50000
#define EE 1600000
#define HH 64
#define F_IN 16
#define N_LAYER_ITERS 3

typedef _Float16 h8  __attribute__((ext_vector_type(8)));
typedef _Float16 h16 __attribute__((ext_vector_type(16)));
typedef float    f8  __attribute__((ext_vector_type(8)));

__device__ __forceinline__ float silu_f(float x) {
    return x / (1.0f + __expf(-x));
}

__device__ __forceinline__ f8 splat8(float x) {
    f8 r;
#pragma unroll
    for (int i = 0; i < 8; ++i) r[i] = x;
    return r;
}

__device__ __forceinline__ f8 wmma16(h16 a, h16 b, f8 c) {
    // D = A(16x32 f16) * B(32x16 f16) + C(16x16 f32)
    return __builtin_amdgcn_wmma_f32_16x16x32_f16(
        false, a, false, b, (short)0, c, false, false);
}

// Load one A-fragment (this lane's 16 f16 of a 16x32 tile):
// lanes 0-15 hold K {0..7,16..23}, lanes 16-31 hold K {8..15,24..31}.
// `base` points at the 32-element K-chunk start for this lane's row; sub = hi?8:0.
__device__ __forceinline__ h16 load_afrag(const _Float16* base, int sub) {
    h8 lo = *(const h8*)(base + sub);
    h8 hi = *(const h8*)(base + sub + 16);
    h16 a;
#pragma unroll
    for (int i = 0; i < 8; ++i) { a[i] = lo[i]; a[i + 8] = hi[i]; }
    return a;
}

__device__ __forceinline__ h16 lds_bfrag(const _Float16* sw, int f, int lane) {
    return *(const h16*)(sw + ((f << 5) + lane) * 16);
}

// ---------------- weight packing: f32 row-major (K x 64) -> per-lane B fragments ----
// layout: P[((c*4 + t)*32 + lane)*16 + j]
__global__ void pack_weights_kernel(const float* __restrict__ W, _Float16* __restrict__ P,
                                    int K, int Kc) {
    int idx = blockIdx.x * blockDim.x + threadIdx.x;
    int total = Kc * 4 * 32 * 16;
    if (idx >= total) return;
    int j    = idx & 15;
    int lane = (idx >> 4) & 31;
    int f    = idx >> 9;
    int t    = f & 3;
    int c    = f >> 2;
    int k = c * 32 + ((j < 8) ? j : j + 8) + ((lane >= 16) ? 8 : 0);
    int n = t * 16 + (lane & 15);
    P[idx] = (k < K) ? (_Float16)W[k * HH + n] : (_Float16)0.0f;
}

// ---------------- misc small kernels ----------------
__global__ void zero_kernel(float* __restrict__ p, int n) {
    int i = blockIdx.x * blockDim.x + threadIdx.x;
    if (i < n) p[i] = 0.0f;
}

__global__ void copy_xv_kernel(const float* __restrict__ xi, const float* __restrict__ vi,
                               float* __restrict__ xo, float* __restrict__ vo) {
    int i = blockIdx.x * blockDim.x + threadIdx.x;
    if (i < NN * 3) { xo[i] = xi[i]; vo[i] = vi[i]; }
}

__global__ void deg_kernel(const int* __restrict__ edges, float* __restrict__ deg) {
    int e = blockIdx.x * blockDim.x + threadIdx.x;
    if (e < EE) atomicAdd(&deg[edges[e]], 1.0f);
}

__global__ void embed_kernel(const float* __restrict__ his, const float* __restrict__ ew,
                             const float* __restrict__ eb, float* __restrict__ h,
                             _Float16* __restrict__ hf) {
    int idx = blockIdx.x * blockDim.x + threadIdx.x;
    if (idx >= NN * HH) return;
    int i = idx >> 6, j = idx & 63;
    float a = eb[j];
#pragma unroll
    for (int k = 0; k < F_IN; ++k) a += his[i * F_IN + k] * ew[k * HH + j];
    h[idx] = a;
    hf[idx] = (_Float16)a;
}

__global__ void out_kernel(const float* __restrict__ x, const float* __restrict__ h,
                           const float* __restrict__ v, float* __restrict__ out) {
    int idx = blockIdx.x * blockDim.x + threadIdx.x;
    if (idx >= NN * (3 + HH + 3)) return;
    if (idx < NN * 3)            out[idx] = x[idx];
    else if (idx < NN * (3 + HH)) out[idx] = h[idx - NN * 3];
    else                          out[idx] = v[idx - NN * (3 + HH)];
}

// ---------------- fused edge kernel (WMMA) ----------------
#define WPB 8   // waves per block (256 threads)

__global__ __launch_bounds__(256)
void edge_kernel(const int* __restrict__ edges, const float* __restrict__ xw,
                 const float* __restrict__ edge_attr, const _Float16* __restrict__ hf,
                 const float* __restrict__ eb1, const float* __restrict__ eb2,
                 const float* __restrict__ cb1, const float* __restrict__ cw2,
                 const _Float16* __restrict__ pw1, const _Float16* __restrict__ pw2,
                 const _Float16* __restrict__ pcw1,
                 float* __restrict__ m_agg, float* __restrict__ coord_agg, int nTiles) {
    // packed weights staged in LDS, shared by all waves of the block
    __shared__ __align__(32) _Float16 sW1[20 * 512];  // edge MLP layer 1 (K=160)
    __shared__ __align__(32) _Float16 sW2[8 * 512];   // edge MLP layer 2 (K=64)
    __shared__ __align__(32) _Float16 sW3[8 * 512];   // coord MLP hidden (K=64)
    __shared__ __align__(32) _Float16 sTile[WPB][16 * 64];  // per-wave transpose tile
    __shared__ int   sRow[WPB][16];
    __shared__ float sG[WPB][16];

    for (int i = threadIdx.x; i < 1280; i += 256) ((uint4*)sW1)[i] = ((const uint4*)pw1)[i];
    for (int i = threadIdx.x; i < 512;  i += 256) ((uint4*)sW2)[i] = ((const uint4*)pw2)[i];
    for (int i = threadIdx.x; i < 512;  i += 256) ((uint4*)sW3)[i] = ((const uint4*)pcw1)[i];
    __syncthreads();

    const int lane = threadIdx.x & 31;
    const int w    = threadIdx.x >> 5;
    const int ln   = lane & 15;
    const bool hi  = lane >= 16;
    const int sub  = hi ? 8 : 0;

    // loop-invariant per-lane biases / coord output weights (n = t*16 + ln)
    float eb1v[4], eb2v[4], cb1v[4], cw2v[4];
#pragma unroll
    for (int t = 0; t < 4; ++t) {
        int n = t * 16 + ln;
        eb1v[t] = eb1[n]; eb2v[t] = eb2[n]; cb1v[t] = cb1[n]; cw2v[t] = cw2[n];
    }

    _Float16* tile = &sTile[w][0];
    const int waveId   = blockIdx.x * WPB + w;
    const int waveStep = gridDim.x * WPB;

    for (int tIdx = waveId; tIdx < nTiles; tIdx += waveStep) {
        const int e = tIdx * 16 + ln;              // lanes l and l+16 share edge l
        const int r = edges[e];
        const int c = edges[EE + e];
        if (!hi) sRow[w][ln] = r;

        const float dx = xw[r * 3 + 0] - xw[c * 3 + 0];
        const float dy = xw[r * 3 + 1] - xw[c * 3 + 1];
        const float dz = xw[r * 3 + 2] - xw[c * 3 + 2];
        const float radial = dx * dx + dy * dy + dz * dz;
        const float ea0 = edge_attr[e * 2 + 0];
        const float ea1 = edge_attr[e * 2 + 1];

        // ---- edge MLP layer 1: ef(160) @ e_w1 -> hidden1 (16x64) ----
        f8 acc[4];
#pragma unroll
        for (int t = 0; t < 4; ++t) acc[t] = splat8(eb1v[t]);

        const _Float16* hr = hf + (size_t)r * HH;
        const _Float16* hc = hf + (size_t)c * HH;
        const _Float16* bases[4] = { hr, hr + 32, hc, hc + 32 };
#pragma unroll
        for (int cc = 0; cc < 4; ++cc) {
            h16 a = load_afrag(bases[cc], sub);
#pragma unroll
            for (int t = 0; t < 4; ++t)
                acc[t] = wmma16(a, lds_bfrag(sW1, cc * 4 + t, lane), acc[t]);
        }
        {   // chunk 4: [radial, ea0, ea1, 0...]; lanes>=16 carry only pad K -> zeros
            h16 a;
#pragma unroll
            for (int i = 0; i < 16; ++i) a[i] = (_Float16)0.0f;
            if (!hi) { a[0] = (_Float16)radial; a[1] = (_Float16)ea0; a[2] = (_Float16)ea1; }
#pragma unroll
            for (int t = 0; t < 4; ++t)
                acc[t] = wmma16(a, lds_bfrag(sW1, 16 + t, lane), acc[t]);
        }
        // SiLU + D-layout -> LDS [edge][feat] f16 (transpose staging)
#pragma unroll
        for (int t = 0; t < 4; ++t)
#pragma unroll
            for (int rr = 0; rr < 8; ++rr) {
                int m = rr + (hi ? 8 : 0), n = t * 16 + ln;
                tile[m * 64 + n] = (_Float16)silu_f(acc[t][rr]);
            }
        asm volatile("" ::: "memory");

        // ---- edge MLP layer 2: hidden1 @ e_w2 -> m (16x64) ----
        h16 a0 = load_afrag(tile + ln * 64 + 0, sub);
        h16 a1 = load_afrag(tile + ln * 64 + 32, sub);
#pragma unroll
        for (int t = 0; t < 4; ++t) {
            acc[t] = splat8(eb2v[t]);
            acc[t] = wmma16(a0, lds_bfrag(sW2, 0 * 4 + t, lane), acc[t]);
            acc[t] = wmma16(a1, lds_bfrag(sW2, 1 * 4 + t, lane), acc[t]);
        }
        // SiLU -> m ; scatter-add m_agg ; restage m into tile for coord MLP
#pragma unroll
        for (int t = 0; t < 4; ++t)
#pragma unroll
            for (int rr = 0; rr < 8; ++rr) {
                int m = rr + (hi ? 8 : 0), n = t * 16 + ln;
                float mv = silu_f(acc[t][rr]);
                atomicAdd(&m_agg[(size_t)sRow[w][m] * HH + n], mv);
                tile[m * 64 + n] = (_Float16)mv;
            }
        asm volatile("" ::: "memory");

        // ---- coord MLP hidden: m @ c_w1, SiLU; then dot with c_w2 -> per-edge gate g
        h16 b0 = load_afrag(tile + ln * 64 + 0, sub);
        h16 b1 = load_afrag(tile + ln * 64 + 32, sub);
        float gsum[8];
#pragma unroll
        for (int rr = 0; rr < 8; ++rr) gsum[rr] = 0.0f;
#pragma unroll
        for (int t = 0; t < 4; ++t) {
            f8 p = splat8(cb1v[t]);
            p = wmma16(b0, lds_bfrag(sW3, 0 * 4 + t, lane), p);
            p = wmma16(b1, lds_bfrag(sW3, 1 * 4 + t, lane), p);
#pragma unroll
            for (int rr = 0; rr < 8; ++rr) gsum[rr] += silu_f(p[rr]) * cw2v[t];
        }
        // reduce across the 16 lanes of each half (n-dimension)
#pragma unroll
        for (int rr = 0; rr < 8; ++rr) {
            gsum[rr] += __shfl_xor(gsum[rr], 1, 32);
            gsum[rr] += __shfl_xor(gsum[rr], 2, 32);
            gsum[rr] += __shfl_xor(gsum[rr], 4, 32);
            gsum[rr] += __shfl_xor(gsum[rr], 8, 32);
        }
        if (ln == 0) {
#pragma unroll
            for (int rr = 0; rr < 8; ++rr) sG[w][rr + (hi ? 8 : 0)] = gsum[rr];
        }
        asm volatile("" ::: "memory");
        if (!hi) {
            float g = sG[w][ln];
            atomicAdd(&coord_agg[r * 3 + 0], dx * g);
            atomicAdd(&coord_agg[r * 3 + 1], dy * g);
            atomicAdd(&coord_agg[r * 3 + 2], dz * g);
        }
    }
}

// ---------------- per-node kernel: vel_scale, v/x update, node MLP, h recurrence ----
__global__ __launch_bounds__(64)
void node_kernel(const float* __restrict__ vw1, const float* __restrict__ vb1,
                 const float* __restrict__ vw2, const float* __restrict__ vb2,
                 const float* __restrict__ nw1, const float* __restrict__ nb1,
                 const float* __restrict__ nw2, const float* __restrict__ nb2,
                 float* __restrict__ h, _Float16* __restrict__ hf,
                 float* __restrict__ x, float* __restrict__ v,
                 const float* __restrict__ m_agg, const float* __restrict__ coord_agg,
                 const float* __restrict__ deg) {
    const int i = blockIdx.x;
    const int j = threadIdx.x;
    __shared__ float hs[64], ms[64], us[64], red[64];

    hs[j] = h[(size_t)i * HH + j];
    __syncthreads();

    // vel_scale = silu(h @ v_w1 + v_b1) @ v_w2 + v_b2
    float a = vb1[j];
#pragma unroll 8
    for (int k = 0; k < HH; ++k) a += hs[k] * vw1[k * HH + j];
    red[j] = silu_f(a) * vw2[j];
    __syncthreads();
    for (int s = 32; s > 0; s >>= 1) {
        if (j < s) red[j] += red[j + s];
        __syncthreads();
    }
    const float vel = red[0] + vb2[0];

    if (j < 3) {
        float d  = fmaxf(deg[i], 1.0f);
        float vo = v[i * 3 + j];
        float vn = vo + coord_agg[i * 3 + j] / d + vel * vo;
        v[i * 3 + j] = vn;
        x[i * 3 + j] += vn;
    }

    ms[j] = m_agg[(size_t)i * HH + j];
    __syncthreads();

    // node MLP: silu(concat(h, m_agg) @ n_w1 + n_b1) @ n_w2 + n_b2
    float a2 = nb1[j];
#pragma unroll 8
    for (int k = 0; k < HH; ++k)
        a2 += hs[k] * nw1[k * HH + j] + ms[k] * nw1[(HH + k) * HH + j];
    us[j] = silu_f(a2);
    __syncthreads();
    float upd = nb2[j];
#pragma unroll 8
    for (int k = 0; k < HH; ++k) upd += us[k] * nw2[k * HH + j];

    // hs_new = h + upd_h ; h = h + hs_new  => 2h + upd
    float hn = 2.0f * hs[j] + upd;
    h[(size_t)i * HH + j]  = hn;
    hf[(size_t)i * HH + j] = (_Float16)hn;
}

// ---------------- host-side launch ----------------
extern "C" void kernel_launch(void* const* d_in, const int* in_sizes, int n_in,
                              void* d_out, int out_size, void* d_ws, size_t ws_size,
                              hipStream_t stream) {
    const float* his  = (const float*)d_in[0];
    const float* x_in = (const float*)d_in[1];
    const float* v_in = (const float*)d_in[2];
    const float* eatt = (const float*)d_in[3];
    const int*   edges = (const int*)d_in[4];
    const float* emb_w = (const float*)d_in[5];
    const float* emb_b = (const float*)d_in[6];
    const float* e_w1 = (const float*)d_in[7];
    const float* e_b1 = (const float*)d_in[8];
    const float* e_w2 = (const float*)d_in[9];
    const float* e_b2 = (const float*)d_in[10];
    const float* c_w1 = (const float*)d_in[11];
    const float* c_b1 = (const float*)d_in[12];
    const float* c_w2 = (const float*)d_in[13];
    const float* n_w1 = (const float*)d_in[14];
    const float* n_b1 = (const float*)d_in[15];
    const float* n_w2 = (const float*)d_in[16];
    const float* n_b2 = (const float*)d_in[17];
    const float* v_w1 = (const float*)d_in[18];
    const float* v_b1 = (const float*)d_in[19];
    const float* v_w2 = (const float*)d_in[20];
    const float* v_b2 = (const float*)d_in[21];

    // workspace layout
    char* ws = (char*)d_ws;
    size_t off = 0;
    auto carve = [&](size_t bytes) { size_t o = off; off = (off + bytes + 255) & ~(size_t)255; return o; };
    float*    h32  = (float*)(ws + carve((size_t)NN * HH * 4));
    _Float16* hf16 = (_Float16*)(ws + carve((size_t)NN * HH * 2));
    float*    xw   = (float*)(ws + carve((size_t)NN * 3 * 4));
    float*    vw   = (float*)(ws + carve((size_t)NN * 3 * 4));
    float*    deg  = (float*)(ws + carve((size_t)NN * 4));
    float*    m_agg = (float*)(ws + carve((size_t)NN * HH * 4));
    float*    c_agg = (float*)(ws + carve((size_t)NN * 3 * 4));
    _Float16* pw1  = (_Float16*)(ws + carve((size_t)20 * 512 * 2));
    _Float16* pw2  = (_Float16*)(ws + carve((size_t)8 * 512 * 2));
    _Float16* pcw1 = (_Float16*)(ws + carve((size_t)8 * 512 * 2));

    const int TPB = 256;
    auto blocks = [&](long n) { return (int)((n + TPB - 1) / TPB); };

    // setup
    zero_kernel<<<blocks(NN), TPB, 0, stream>>>(deg, NN);
    deg_kernel<<<blocks(EE), TPB, 0, stream>>>(edges, deg);
    copy_xv_kernel<<<blocks(NN * 3), TPB, 0, stream>>>(x_in, v_in, xw, vw);
    embed_kernel<<<blocks((long)NN * HH), TPB, 0, stream>>>(his, emb_w, emb_b, h32, hf16);
    pack_weights_kernel<<<blocks(20 * 512), TPB, 0, stream>>>(e_w1, pw1, 2 * HH + 3, 5);
    pack_weights_kernel<<<blocks(8 * 512), TPB, 0, stream>>>(e_w2, pw2, HH, 2);
    pack_weights_kernel<<<blocks(8 * 512), TPB, 0, stream>>>(c_w1, pcw1, HH, 2);

    const int nTiles = EE / 16;
    const int edgeBlocks = 2048;

    for (int layer = 0; layer < N_LAYER_ITERS; ++layer) {
        zero_kernel<<<blocks((long)NN * HH), TPB, 0, stream>>>(m_agg, NN * HH);
        zero_kernel<<<blocks(NN * 3), TPB, 0, stream>>>(c_agg, NN * 3);
        edge_kernel<<<edgeBlocks, 256, 0, stream>>>(
            edges, xw, eatt, hf16, e_b1, e_b2, c_b1, c_w2,
            pw1, pw2, pcw1, m_agg, c_agg, nTiles);
        node_kernel<<<NN, 64, 0, stream>>>(
            v_w1, v_b1, v_w2, v_b2, n_w1, n_b1, n_w2, n_b2,
            h32, hf16, xw, vw, m_agg, c_agg, deg);
    }

    out_kernel<<<blocks((long)NN * (3 + HH + 3)), TPB, 0, stream>>>(xw, h32, vw, (float*)d_out);
}
